// BesselConv2d_34746285425259
// MI455X (gfx1250) — matrix operand order repaired and checked by reading, not compiled, not run on hardware
//
#include <hip/hip_runtime.h>

// ---------------- problem constants ----------------
#define KS   9
#define KK   81            // KS*KS
#define MM   16            // angular freqs
#define JJ   8             // radial orders
#define CI   64
#define CO   64
#define NB   32
#define HH   64
#define WW   64
#define PADR 4
#define HP   72            // HH + 2*PADR
#define WP   72
#define KTOT (KK * CI)     // 5184  (K-order: (kh*9+kw)*64 + ci)
#define NGRP 32            // 2 * MM  (real/imag x m)
#define CC   (CI * CO)     // 4096

#define XPAD_ELEMS ((long)NB * HP * WP * CI)   // 10,616,832 bf16
#define FILT_ELEMS ((long)NGRP * CO * KTOT)    // 10,616,832 bf16

typedef __bf16 bf16_t;
typedef __attribute__((ext_vector_type(16))) __bf16 v16bf;
typedef __attribute__((ext_vector_type(8)))  __bf16 v8bf;
typedef __attribute__((ext_vector_type(8)))  float   v8f;

// ---------------------------------------------------------------------------
// Stage 1: x (N,CI,H,W) f32  ->  xpad (N,HP,WP,CI) bf16 with zero halo.
// NHWC so a 32-ci slab at one (kh,kw) tap is contiguous: ideal WMMA B frags,
// and the pre-zeroed halo removes all bounds checks from the hot loop.
// ---------------------------------------------------------------------------
__global__ void bessel_xpad_kernel(const float* __restrict__ x,
                                   bf16_t* __restrict__ xpad) {
    long idx = (long)blockIdx.x * blockDim.x + threadIdx.x;
    if (idx >= XPAD_ELEMS) return;
    int ci = (int)(idx & 63);
    long t = idx >> 6;
    int wp = (int)(t % WP); t /= WP;
    int hp = (int)(t % HP); t /= HP;
    int n  = (int)t;
    int h = hp - PADR, w = wp - PADR;
    float v = 0.0f;
    if (h >= 0 && h < HH && w >= 0 && w < WW)
        v = x[(((long)n * CI + ci) * HH + h) * WW + w];
    xpad[idx] = (bf16_t)v;
}

// ---------------------------------------------------------------------------
// Stage 2: build spectral->spatial filter bank in bf16.
//   p=0 (real): sum_j Tr*w_r - Ti*w_i ; p=1 (imag): sum_j Tr*w_i + Ti*w_r
// Layout: filt[(p*16+m)*64+co][(kh*9+kw)*64+ci]  -> flat index == thread idx.
// Negligible FLOPs (0.34 GF); T/w arrays are L2-resident.
// ---------------------------------------------------------------------------
__global__ void bessel_filt_kernel(const float* __restrict__ Tr,
                                   const float* __restrict__ Ti,
                                   const float* __restrict__ wr,
                                   const float* __restrict__ wi,
                                   bf16_t* __restrict__ filt) {
    long idx = (long)blockIdx.x * blockDim.x + threadIdx.x;
    if (idx >= FILT_ELEMS) return;
    int ci  = (int)(idx & 63);
    long t  = idx >> 6;
    int khw = (int)(t % KK); t /= KK;
    int co  = (int)(t & 63); t >>= 6;
    int m   = (int)(t & 15);
    int p   = (int)(t >> 4);

    const float* Ta = Tr + ((long)m * KK + khw) * JJ;
    const float* Tb = Ti + ((long)m * KK + khw) * JJ;
    long woff = (long)m * JJ * CC + (long)ci * CO + co;   // [m][j][ci*CO+co]
    const float* wa = (p == 0 ? wr : wi) + woff;
    const float* wb = (p == 0 ? wi : wr) + woff;
    float s = (p == 0) ? -1.0f : 1.0f;

    float acc = 0.0f;
#pragma unroll
    for (int j = 0; j < JJ; ++j)
        acc += Ta[j] * wa[(long)j * CC] + s * Tb[j] * wb[(long)j * CC];
    filt[idx] = (bf16_t)acc;
}

// ---------------------------------------------------------------------------
// Stage 3: fused implicit-GEMM conv + square + group-reduce + bias.
// Grid: one block per (n, h-row-pair) -> 1024 blocks, 256 threads = 8 waves.
// Block tile: 64 cout x 128 positions (2 rows x 64 w).
// Wave tile (2x2): 2 co-blocks x 2 pos-blocks -> per 32-K step it loads
// 2 A frags + 2 B frags (8x b128) and issues 4 WMMAs (2 loads/WMMA), with
// A frags reused across pos tiles and B frags across co tiles in registers.
// For each of the 32 (part,m) groups the K=5184 GEMM is run in 32-K steps
// (one (kh,kw) tap, 32 ci); the f32 C-tiles are squared and accumulated.
// Finally bias is added and out (NCHW f32) written exactly once per element.
// ---------------------------------------------------------------------------
__global__ __launch_bounds__(256)
void bessel_conv_kernel(const bf16_t* __restrict__ xpad,
                        const bf16_t* __restrict__ filt,
                        const float* __restrict__ bias,
                        float* __restrict__ out) {
    const int blk = blockIdx.x;          // 0..1023
    const int n  = blk >> 5;
    const int h0 = (blk & 31) * 2;       // two output rows: h0, h0+1

    const int lane = threadIdx.x & 31;
    const int wv   = threadIdx.x >> 5;   // 0..7
    const int ch   = wv & 1;             // co half: blocks {2ch, 2ch+1}
    const int pq   = wv >> 1;            // 0..3 position quarter
    const int row  = pq >> 1;            // which of the two h rows
    const int ws0  = (pq & 1) * 32;      // first of two adjacent 16-w segments

    const int col  = lane & 15;
    const int hi   = lane >> 4;          // 0 or 1

    // Per-lane A bases: rows (cb*16 + col) of the filter bank, +16 K for hi lanes.
    const bf16_t* Abase0 = filt + (long)((2 * ch) * 16 + col) * KTOT + hi * 16;
    const bf16_t* Abase1 = Abase0 + (long)16 * KTOT;
    const bf16_t* xim = xpad + ((long)n * HP + (h0 + row)) * WP * CI;

    v8f oa00 = {}; v8f oa01 = {}; v8f oa10 = {}; v8f oa11 = {};

    for (int g = 0; g < NGRP; ++g) {
        v8f c00 = {}; v8f c01 = {}; v8f c10 = {}; v8f c11 = {};
        const long goff = (long)g * CO * KTOT;
        const bf16_t* Ag0 = Abase0 + goff;
        const bf16_t* Ag1 = Abase1 + goff;

        for (int kh = 0; kh < KS; ++kh) {
            const bf16_t* xrow = xim + (long)kh * WP * CI;
#pragma unroll
            for (int kw = 0; kw < KS; ++kw) {
                // B columns: output w = ws0 + {0..15} and ws0+16 + {0..15}
                const bf16_t* xb0 = xrow + (long)(ws0 + col + kw) * CI + hi * 16;
                const bf16_t* xb1 = xb0 + 16 * CI;
#pragma unroll
                for (int ci0 = 0; ci0 < CI; ci0 += 32) {
                    const int k0 = (kh * KS + kw) * CI + ci0;

                    // A fragments (16x32 bf16): K {k0..+15}
                    // (+16 folded in for hi lanes).
                    const bf16_t* ap0 = Ag0 + k0;
                    const bf16_t* ap1 = Ag1 + k0;
                    v8bf a0l = *(const v8bf*)(ap0);
                    v8bf a0h = *(const v8bf*)(ap0 + 8);
                    v8bf a1l = *(const v8bf*)(ap1);
                    v8bf a1h = *(const v8bf*)(ap1 + 8);
                    v16bf a0 = __builtin_shufflevector(a0l, a0h,
                        0, 1, 2, 3, 4, 5, 6, 7, 8, 9, 10, 11, 12, 13, 14, 15);
                    v16bf a1 = __builtin_shufflevector(a1l, a1h,
                        0, 1, 2, 3, 4, 5, 6, 7, 8, 9, 10, 11, 12, 13, 14, 15);

                    // keep the streaming filter rows flowing toward L0
                    __builtin_prefetch(ap0 + 256, 0, 1);
                    __builtin_prefetch(ap1 + 256, 0, 1);

                    // B fragments (32x16 bf16): 16 consecutive ci per lane.
                    v16bf b0 = *(const v16bf*)(xb0 + ci0);
                    v16bf b1 = *(const v16bf*)(xb1 + ci0);

                    c00 = __builtin_amdgcn_wmma_f32_16x16x32_bf16(
                            false, a0, false, b0, (short)0, c00, false, false);
                    c01 = __builtin_amdgcn_wmma_f32_16x16x32_bf16(
                            false, a0, false, b1, (short)0, c01, false, false);
                    c10 = __builtin_amdgcn_wmma_f32_16x16x32_bf16(
                            false, a1, false, b0, (short)0, c10, false, false);
                    c11 = __builtin_amdgcn_wmma_f32_16x16x32_bf16(
                            false, a1, false, b1, (short)0, c11, false, false);
                }
            }
        }
#pragma unroll
        for (int i = 0; i < 8; ++i) {
            oa00[i] += c00[i] * c00[i];
            oa01[i] += c01[i] * c01[i];
            oa10[i] += c10[i] * c10[i];
            oa11[i] += c11[i] * c11[i];
        }
    }

    // Store: C VGPR v -> co = cb*16 + v + 8*hi ; column -> w = wseg + col.
    const int h = h0 + row;
#pragma unroll
    for (int v = 0; v < 8; ++v) {
        const int co0 = (2 * ch) * 16 + v + hi * 8;
        const int co1 = co0 + 16;
        const float bc0 = bias[co0];
        const float bc1 = bias[co1];
        const long base0 = (((long)n * CO + co0) * HH + h) * WW;
        const long base1 = (((long)n * CO + co1) * HH + h) * WW;
        out[base0 + ws0 + col]      = oa00[v] + bc0;
        out[base0 + ws0 + 16 + col] = oa01[v] + bc0;
        out[base1 + ws0 + col]      = oa10[v] + bc1;
        out[base1 + ws0 + 16 + col] = oa11[v] + bc1;
    }
}

// ---------------------------------------------------------------------------
extern "C" void kernel_launch(void* const* d_in, const int* in_sizes, int n_in,
                              void* d_out, int out_size, void* d_ws, size_t ws_size,
                              hipStream_t stream) {
    const float* x  = (const float*)d_in[0];
    const float* Tr = (const float*)d_in[1];
    const float* Ti = (const float*)d_in[2];
    const float* wr = (const float*)d_in[3];
    const float* wi = (const float*)d_in[4];
    const float* b  = (const float*)d_in[5];
    float* out = (float*)d_out;

    bf16_t* xpad = (bf16_t*)d_ws;
    bf16_t* filt = (bf16_t*)((char*)d_ws + XPAD_ELEMS * sizeof(bf16_t));

    const int thr = 256;
    const int gx_pad  = (int)((XPAD_ELEMS + thr - 1) / thr);   // 41472
    const int gx_filt = (int)((FILT_ELEMS + thr - 1) / thr);   // 41472

    bessel_xpad_kernel<<<gx_pad, thr, 0, stream>>>(x, xpad);
    bessel_filt_kernel<<<gx_filt, thr, 0, stream>>>(Tr, Ti, wr, wi, filt);
    bessel_conv_kernel<<<NB * (HH / 2), thr, 0, stream>>>(xpad, filt, b, out);
}
